// B_M_68831145885709
// MI455X (gfx1250) — matrix operand order repaired and checked by reading, not compiled, not run on hardware
//
#include <hip/hip_runtime.h>
#include <hip/hip_bf16.h>

// ---------------------------------------------------------------------------
// LIIF-style fused MLP for MI455X (gfx1250), wave32 + WMMA f16 (f32 accum).
//
// Row space: 262,144 rows = (b*Q+q)*4 + shift. 128 rows / workgroup,
// 8 waves x 16 rows. All five GEMMs fused; weights pre-packed to f16
// WMMA B-fragment layout in d_ws by pack_weights(). Weight panels are
// double-buffered in LDS (at offset 0 so every fragment address fits the
// 16-bit DS immediate) and staged with the CDNA5 async global->LDS path
// (GLOBAL_LOAD_ASYNC_TO_LDS_B128 / ASYNCcnt).
// ---------------------------------------------------------------------------

typedef _Float16 h8   __attribute__((ext_vector_type(8)));
typedef _Float16 v16h __attribute__((ext_vector_type(16)));
typedef float    v8f  __attribute__((ext_vector_type(8)));

// GCC-style vector type matching the async-LDS builtin's parameter type.
typedef int v4i __attribute__((__vector_size__(16)));
typedef __attribute__((address_space(1))) v4i* gv4i_ptr;   // global (AS1)
typedef __attribute__((address_space(3))) v4i* lv4i_ptr;   // LDS (AS3)

#if __has_builtin(__builtin_amdgcn_global_load_async_to_lds_b128)
#define HAVE_ASYNC_LDS 1
#else
#define HAVE_ASYNC_LDS 0
#endif

__device__ __forceinline__ v16h make_frag(const _Float16* p0, const _Float16* p1) {
    h8 lo = *(const h8*)p0;
    h8 hi = *(const h8*)p1;
    return __builtin_shufflevector(lo, hi, 0, 1, 2, 3, 4, 5, 6, 7,
                                           8, 9, 10, 11, 12, 13, 14, 15);
}

// Stage n16*16 bytes of packed weights into an LDS panel.
__device__ __forceinline__ void stage_panel(const _Float16* __restrict__ src,
                                            _Float16* __restrict__ dst,
                                            int n16, int tid) {
    const uint4* s4 = (const uint4*)src;
    uint4* d4 = (uint4*)dst;
#if HAVE_ASYNC_LDS
    for (int i = tid; i < n16; i += 256)
        __builtin_amdgcn_global_load_async_to_lds_b128(
            (gv4i_ptr)(s4 + i), (lv4i_ptr)(d4 + i), 0, 0);
#else
    for (int i = tid; i < n16; i += 256) d4[i] = s4[i];
#endif
}

__device__ __forceinline__ void wait_async() {
#if HAVE_ASYNC_LDS
#if __has_builtin(__builtin_amdgcn_s_wait_asynccnt)
    __builtin_amdgcn_s_wait_asynccnt(0);
#else
    asm volatile("s_wait_asynccnt 0x0" ::: "memory");
#endif
#endif
}

// ---------------------------------------------------------------------------
// Weight packing: fp32 row-major [K x N] -> f16 WMMA B fragments.
// Fragment (kt, nt) = 1024 bytes: 32 lanes x 16 halves.
//   lane (lh=lane>>4, nl=lane&15): half j -> W[kt*32 + lh*16 + j][nt*16 + nl]
// Packed half offsets: L1:0  L2:16384  L3:81920  L4:147456  L5:212992
// ---------------------------------------------------------------------------
__global__ __launch_bounds__(256) void pack_weights(
    const float* __restrict__ W1, const float* __restrict__ W2,
    const float* __restrict__ W3, const float* __restrict__ W4,
    const float* __restrict__ W5, _Float16* __restrict__ wpack)
{
    const int idx = blockIdx.x * 256 + threadIdx.x;
    if (idx >= 221184) return;

    int off, ntl, kact, nact;
    const float* W;
    if (idx < 16384)       { off = 0;      ntl = 16; kact = 40;  nact = 256; W = W1; }
    else if (idx < 81920)  { off = 16384;  ntl = 16; kact = 256; nact = 256; W = W2; }
    else if (idx < 147456) { off = 81920;  ntl = 16; kact = 256; nact = 256; W = W3; }
    else if (idx < 212992) { off = 147456; ntl = 16; kact = 256; nact = 256; W = W4; }
    else                   { off = 212992; ntl = 2;  kact = 256; nact = 31;  W = W5; }

    const int t      = idx - off;
    const int frag   = t >> 9;        // 512 halves per fragment
    const int within = t & 511;
    const int lane   = within >> 4;
    const int j      = within & 15;
    const int kt     = frag / ntl;
    const int nt     = frag % ntl;
    const int K      = kt * 32 + (lane >> 4) * 16 + j;
    const int n      = nt * 16 + (lane & 15);
    const float v = (K < kact && n < nact) ? W[K * nact + n] : 0.0f;
    wpack[idx] = (_Float16)v;
}

// ---------------------------------------------------------------------------
// One MLP layer: act[16x256 f16, per wave] @ W[K x N] -> act (bias+ReLU).
// KT = K/32 chunks, NTL = N/16 tiles, PKT = kt-chunks per LDS panel.
// Weight panels double-buffered: async-stage p+1 while computing p.
// wbuf sits at LDS offset 0 => all fragment addresses = lane base + imm16.
// ---------------------------------------------------------------------------
template <int KT, int NTL, int PKT>
__device__ __forceinline__ void mlp_layer(
    const _Float16* __restrict__ wsrc, const float* __restrict__ bias,
    _Float16* __restrict__ act, _Float16* __restrict__ wbuf,
    int tid, int lane, int lh, int nl)
{
    constexpr int NPAN    = KT / PKT;
    constexpr int PHALVES = PKT * NTL * 512;      // halves per panel
    constexpr int PN16    = PHALVES / 8;          // 16B chunks per panel

    v8f acc[NTL];
#pragma unroll
    for (int n = 0; n < NTL; ++n)
#pragma unroll
        for (int i = 0; i < 8; ++i) acc[n][i] = 0.0f;

    // per-lane fragment base: wbuf + lane*32 bytes; everything else is imm16
    const _Float16* wlane = wbuf + lane * 16;

    // prologue: stage panel 0 into buffer 0
    __syncthreads();
    stage_panel(wsrc, wbuf, PN16, tid);
    wait_async();
    __syncthreads();

#pragma unroll
    for (int p = 0; p < NPAN; ++p) {
        // overlap: issue async stage of panel p+1 into the other buffer
        if (p + 1 < NPAN)
            stage_panel(wsrc + (size_t)(p + 1) * PHALVES,
                        wbuf + (size_t)((p + 1) & 1) * PHALVES, PN16, tid);

#pragma unroll
        for (int kl = 0; kl < PKT; ++kl) {
            const int kt = p * PKT + kl;
            // A fragment: lane (nl) = row M, halves K = kt*32 + lh*16 + j
            const _Float16* arow = act + nl * 256 + kt * 32 + lh * 16;
            const v16h afrag = make_frag(arow, arow + 8);
#pragma unroll
            for (int n = 0; n < NTL; ++n) {
                // byte offset = panel*32768 + (kl*NTL+n)*1024  (< 64KB)
                const _Float16* bp = wlane + ((p & 1) ? PHALVES : 0)
                                           + (kl * NTL + n) * 512;
                const v16h bfrag = make_frag(bp, bp + 8);
                acc[n] = __builtin_amdgcn_wmma_f32_16x16x32_f16(
                    false, afrag, false, bfrag, (short)0, acc[n], false, false);
            }
        }
        wait_async();        // panel p+1 resident in LDS for this wave
        __syncthreads();     // all waves: p consumed, p+1 visible
    }

    // epilogue: bias + ReLU, write back as f16 (all K already consumed)
#pragma unroll
    for (int n = 0; n < NTL; ++n) {
        const int col = n * 16 + nl;
        const float bv = bias[col];
#pragma unroll
        for (int i = 0; i < 8; ++i) {
            float v = acc[n][i] + bv;
            v = fmaxf(v, 0.0f);
            const int m = i + lh * 8;     // D layout: M = vgpr + lh*8, N = nl
            act[m * 256 + col] = (_Float16)v;
        }
    }
}

// ---------------------------------------------------------------------------
// Fused kernel: gather -> 5-layer MLP -> area-weighted 4-tap blend.
// ---------------------------------------------------------------------------
__global__ __launch_bounds__(256) void liif_mlp(
    const float* __restrict__ feat, const float* __restrict__ coord,
    const float* __restrict__ cell,
    const float* __restrict__ b1, const float* __restrict__ b2,
    const float* __restrict__ b3, const float* __restrict__ b4,
    const float* __restrict__ b5,
    const _Float16* __restrict__ wpack, float* __restrict__ out)
{
    extern __shared__ char smem[];
    _Float16* wbuf   = (_Float16*)smem;                  // 2 x 32KB weight panels @ 0
    _Float16* actAll = (_Float16*)(smem + 65536);        // 8 waves * 16x256 f16 = 64KB
    float*    areas  = (float*)(smem + 65536 + 65536);   // 8*16 floats

    const int tid  = threadIdx.x;
    const int wave = tid >> 5;
    const int lane = tid & 31;
    const int lh   = lane >> 4;
    const int nl   = lane & 15;
    _Float16* act  = actAll + wave * 4096;
    float* area_w  = areas + wave * 16;

    // ---- Phase 0: build 40-dim inputs; lanes 0..15 each own one row ----
    if (lane < 16) {
        const int m    = lane;
        const int row  = blockIdx.x * 128 + wave * 16 + m;
        const int sidx = row >> 2;             // b*Q + q   (Q = 32768)
        const int s    = row & 3;
        const int b    = sidx >> 15;
        const float cy = coord[sidx * 2 + 0];
        const float cx = coord[sidx * 2 + 1];
        const float r  = 1.0f / 64.0f;
        const float sy = ((s & 2) ? r : -r) + 1e-6f;
        const float sx = ((s & 1) ? r : -r) + 1e-6f;
        const float c0 = fminf(fmaxf(cy + sy, -1.0f + 1e-6f), 1.0f - 1e-6f);
        const float c1 = fminf(fmaxf(cx + sx, -1.0f + 1e-6f), 1.0f - 1e-6f);
        int iy = (int)floorf(((c0 + 1.0f) * 64.0f - 1.0f) * 0.5f + 0.5f);
        int ix = (int)floorf(((c1 + 1.0f) * 64.0f - 1.0f) * 0.5f + 0.5f);
        iy = min(max(iy, 0), 63);
        ix = min(max(ix, 0), 63);
        const float qy   = -1.0f + (2.0f * (float)iy + 1.0f) * r;
        const float qx   = -1.0f + (2.0f * (float)ix + 1.0f) * r;
        const float rely = (cy - qy) * 64.0f;
        const float relx = (cx - qx) * 64.0f;
        const float cly  = cell[sidx * 2 + 0] * 64.0f;
        const float clx  = cell[sidx * 2 + 1] * 64.0f;
        const float* fb  = feat + b * (4 * 64 * 64);
#pragma unroll
        for (int c = 0; c < 4; ++c)
#pragma unroll
            for (int ki = 0; ki < 3; ++ki)
#pragma unroll
                for (int kj = 0; kj < 3; ++kj) {
                    const int yy = iy + ki - 1, xx = ix + kj - 1;
                    const bool ok = (yy >= 0) & (yy < 64) & (xx >= 0) & (xx < 64);
                    const float v = ok ? fb[(c * 64 + yy) * 64 + xx] : 0.0f;
                    act[m * 256 + c * 9 + ki * 3 + kj] = (_Float16)v;
                }
        act[m * 256 + 36] = (_Float16)rely;
        act[m * 256 + 37] = (_Float16)relx;
        act[m * 256 + 38] = (_Float16)cly;
        act[m * 256 + 39] = (_Float16)clx;
#pragma unroll
        for (int k = 40; k < 64; ++k) act[m * 256 + k] = (_Float16)0.0f;
        area_w[m] = fabsf(rely * relx) + 1e-9f;
    }

    // ---- Layers 1..4 (K pad 64 / 256, N = 256, ReLU) ----
    mlp_layer<2, 16, 2>(wpack + 0,      b1, act, wbuf, tid, lane, lh, nl);
    mlp_layer<8, 16, 2>(wpack + 16384,  b2, act, wbuf, tid, lane, lh, nl);
    mlp_layer<8, 16, 2>(wpack + 81920,  b3, act, wbuf, tid, lane, lh, nl);
    mlp_layer<8, 16, 2>(wpack + 147456, b4, act, wbuf, tid, lane, lh, nl);

    // ---- Layer 5 (K=256, N pad 32) + 4-shift area blend ----
    v8f acc[2];
#pragma unroll
    for (int n = 0; n < 2; ++n)
#pragma unroll
        for (int i = 0; i < 8; ++i) acc[n][i] = 0.0f;

    __syncthreads();
    stage_panel(wpack + 212992, wbuf, 1024, tid);   // 16KB of layer-5 fragments
    wait_async();
    __syncthreads();

    const _Float16* wlane = wbuf + lane * 16;
#pragma unroll
    for (int kt = 0; kt < 8; ++kt) {
        const _Float16* arow = act + nl * 256 + kt * 32 + lh * 16;
        const v16h afrag = make_frag(arow, arow + 8);
#pragma unroll
        for (int n = 0; n < 2; ++n) {
            const _Float16* bp = wlane + (kt * 2 + n) * 512;
            const v16h bfrag = make_frag(bp, bp + 8);
            acc[n] = __builtin_amdgcn_wmma_f32_16x16x32_f16(
                false, afrag, false, bfrag, (short)0, acc[n], false, false);
        }
    }

    // acc element i holds row m = i + lh*8 of this wave's 16 rows:
    //   g = i>>2 , s = i&3 , local sample t = g + 2*lh  -> blend within lane.
#pragma unroll
    for (int nt = 0; nt < 2; ++nt) {
        const int col = nt * 16 + nl;
        if (col < 31) {
            const float bv = b5[col];
#pragma unroll
            for (int g = 0; g < 2; ++g) {
                const int t = g + lh * 2;
                const float a0 = area_w[t * 4 + 0];
                const float a1 = area_w[t * 4 + 1];
                const float a2 = area_w[t * 4 + 2];
                const float a3 = area_w[t * 4 + 3];
                const float inv = 1.0f / (a0 + a1 + a2 + a3);
                const float v =
                    ((acc[nt][g * 4 + 0] + bv) * a3 +
                     (acc[nt][g * 4 + 1] + bv) * a2 +
                     (acc[nt][g * 4 + 2] + bv) * a1 +
                     (acc[nt][g * 4 + 3] + bv) * a0) * inv;
                const int gs = blockIdx.x * 32 + wave * 4 + t;   // global sample
                out[gs * 31 + col] = v;
            }
        }
    }
}

// ---------------------------------------------------------------------------
extern "C" void kernel_launch(void* const* d_in, const int* in_sizes, int n_in,
                              void* d_out, int out_size, void* d_ws, size_t ws_size,
                              hipStream_t stream) {
    (void)in_sizes; (void)n_in; (void)out_size; (void)ws_size;
    const float* feat  = (const float*)d_in[0];
    const float* coord = (const float*)d_in[1];
    const float* cell  = (const float*)d_in[2];
    const float* W1 = (const float*)d_in[3];  const float* b1 = (const float*)d_in[4];
    const float* W2 = (const float*)d_in[5];  const float* b2 = (const float*)d_in[6];
    const float* W3 = (const float*)d_in[7];  const float* b3 = (const float*)d_in[8];
    const float* W4 = (const float*)d_in[9];  const float* b4 = (const float*)d_in[10];
    const float* W5 = (const float*)d_in[11]; const float* b5 = (const float*)d_in[12];

    _Float16* wpack = (_Float16*)d_ws;                 // 221184 halves = 432KB
    float* out = (float*)d_out;

    const int SMEM_BYTES = 65536 + 65536 + 512;        // 128.5KB of 320KB/WGP

    hipError_t attr_err = hipFuncSetAttribute(
        (const void*)liif_mlp, hipFuncAttributeMaxDynamicSharedMemorySize, SMEM_BYTES);
    (void)attr_err;

    pack_weights<<<864, 256, 0, stream>>>(W1, W2, W3, W4, W5, wpack);
    liif_mlp<<<2048, 256, SMEM_BYTES, stream>>>(feat, coord, cell,
                                                b1, b2, b3, b4, b5, wpack, out);
}
